// BipartiteGCN_52553219834468
// MI455X (gfx1250) — matrix operand order repaired and checked by reading, not compiled, not run on hardware
//
#include <hip/hip_runtime.h>
#include <hip/hip_bf16.h>

typedef _Float16 h16;
typedef __attribute__((ext_vector_type(2)))  _Float16 v2h;
typedef __attribute__((ext_vector_type(8)))  _Float16 v8h;
typedef __attribute__((ext_vector_type(16))) _Float16 v16h;
typedef __attribute__((ext_vector_type(8)))  float    v8f;

#define EPSF 1e-5f

// ---------------- device helpers ----------------

template<int ACT>
__device__ __forceinline__ float apply_act(float v) {
    if (ACT == 1) return fmaxf(v, 0.f);
    if (ACT == 2) return v > 0.f ? v : 0.01f * v;
    return v;
}

// A fragment: 16x32 f16 tile of X (row-major, leading dim = K).
// ISA layout: lanes 0-15 hold M=0..15 with K=[0..7,16..23]; lanes 16-31 K=[8..15,24..31].
__device__ __forceinline__ v16h load_a_frag(const h16* __restrict__ X, int ldx,
                                            long long row0, int k0, int lane) {
    int m  = lane & 15;
    int hi = lane >> 4;
    const h16* p = X + (size_t)(row0 + m) * ldx + k0 + hi * 8;
    v8h lo = *(const v8h*)(p);
    v8h up = *(const v8h*)(p + 16);
    return __builtin_shufflevector(lo, up, 0,1,2,3,4,5,6,7,8,9,10,11,12,13,14,15);
}

// B fragment: 32x16 f16 tile of W (K x 64), staged transposed in LDS as wt[n*K + k].
// ISA layout: lanes 0-15 hold K=[0..7,16..23] for N=lane; lanes 16-31 hold K=[8..15,24..31].
__device__ __forceinline__ v16h load_b_frag(const h16* wt, int K,
                                            int n0, int k0, int lane) {
    int n  = n0 + (lane & 15);
    int hi = lane >> 4;
    const h16* p = wt + (size_t)n * K + k0 + hi * 8;
    v8h lo = *(const v8h*)(p);
    v8h up = *(const v8h*)(p + 16);
    return __builtin_shufflevector(lo, up, 0,1,2,3,4,5,6,7,8,9,10,11,12,13,14,15);
}

// ---------------- kernels ----------------

// LayerNorm(NF) -> Linear(NF,64) -> LeakyReLU, output f16. Thread per (node, col).
__global__ void embed_front_kernel(const float* __restrict__ x,
                                   const float* __restrict__ lnw, const float* __restrict__ lnb,
                                   const float* __restrict__ W1,  const float* __restrict__ b1,
                                   h16* __restrict__ out, int N, int NF) {
    int i = blockIdx.x * blockDim.x + threadIdx.x;
    if (i >= N * 64) return;
    int node = i >> 6, col = i & 63;
    const float* xr = x + (size_t)node * NF;
    float m = 0.f;
    for (int f = 0; f < NF; ++f) m += xr[f];
    m /= (float)NF;
    float v = 0.f;
    for (int f = 0; f < NF; ++f) { float d = xr[f] - m; v += d * d; }
    v /= (float)NF;
    float rs = rsqrtf(v + EPSF);
    float acc = b1[col];
    for (int f = 0; f < NF; ++f) {
        float xn = (xr[f] - m) * rs * lnw[f] + lnb[f];
        acc += xn * W1[f * 64 + col];
    }
    acc = acc > 0.f ? acc : 0.01f * acc;
    out[i] = (h16)acc;
}

// Y[rows x 64] = act(X[rows x K] @ W[K x 64] + bias), X/Y f16, W/bias f32.
// W staged transposed to LDS once per block; B fragments hoisted to registers
// once per wave; wave grid-strides over 16-row tiles (steady state: A loads + WMMA).
// launch_bounds(256,1): allow full VGPR budget -- B fragments + accs must stay
// register-resident (no scratch spills in the inner loop).
template<int K, int ACT>
__global__ void __launch_bounds__(256, 1)
gemm_tile_kernel(const h16* __restrict__ X, const float* __restrict__ W,
                 const float* __restrict__ bias, h16* __restrict__ Y,
                 int rows) {
    __shared__ alignas(64) h16 wt[64 * K];
    __shared__ float bl[64];
    int tid = threadIdx.x;
    for (int i = tid; i < K * 64; i += 256) {
        int k = i >> 6, n = i & 63;          // W is [K x 64] row-major
        wt[n * K + k] = (h16)W[i];
    }
    if (tid < 64) bl[tid] = bias ? bias[tid] : 0.f;
    __syncthreads();

    int lane = tid & 31, wv = tid >> 5;
    constexpr int KC = K / 32;

    v16h bfrag[KC][4];
#pragma unroll
    for (int kc = 0; kc < KC; ++kc)
#pragma unroll
        for (int nt = 0; nt < 4; ++nt)
            bfrag[kc][nt] = load_b_frag(wt, K, nt * 16, kc * 32, lane);

    int n_ = lane & 15, hi = lane >> 4;
    float bb[4];
#pragma unroll
    for (int nt = 0; nt < 4; ++nt) bb[nt] = bl[nt * 16 + n_];

    int tiles = (rows + 15) / 16;
    for (int tile = blockIdx.x * 8 + wv; tile < tiles; tile += gridDim.x * 8) {
        int row0 = tile * 16;
        v8f acc[4] = {};
#pragma unroll
        for (int kc = 0; kc < KC; ++kc) {
            v16h a = load_a_frag(X, K, row0, kc * 32, lane);
#pragma unroll
            for (int nt = 0; nt < 4; ++nt) {
                acc[nt] = __builtin_amdgcn_wmma_f32_16x16x32_f16(
                    false, a, false, bfrag[kc][nt], (short)0, acc[nt], false, false);
            }
        }
#pragma unroll
        for (int nt = 0; nt < 4; ++nt) {
            int n = nt * 16 + n_;
#pragma unroll
            for (int r = 0; r < 8; ++r) {
                float v = apply_act<ACT>(acc[nt][r] + bb[nt]);
                Y[(size_t)(row0 + r + 8 * hi) * 64 + n] = (h16)v;
            }
        }
    }
}

__global__ void zero_f32_kernel(float* __restrict__ p, long long n) {
    long long i = (long long)blockIdx.x * blockDim.x + threadIdx.x;
    if (i < n) p[i] = 0.f;
}

// T[e] = relu(layernorm(h_i[tgt[e]] + h_j[src[e]]))  -- wave per edge, 2 cols/lane.
__global__ void edge_prep_kernel(const h16* __restrict__ h_i, const h16* __restrict__ h_j,
                                 const long long* __restrict__ tgt,
                                 const long long* __restrict__ src,
                                 const float* __restrict__ flnw, const float* __restrict__ flnb,
                                 h16* __restrict__ T, long long E) {
    int tid = threadIdx.x, lane = tid & 31, wv = tid >> 5;
    long long e = (long long)blockIdx.x * 8 + wv;
    if (e >= E) return;
    long long t = tgt[e], s = src[e];
    int c0 = lane * 2;
    v2h a = *(const v2h*)(h_i + (size_t)t * 64 + c0);
    v2h b = *(const v2h*)(h_j + (size_t)s * 64 + c0);
    float x0 = (float)a[0] + (float)b[0];
    float x1 = (float)a[1] + (float)b[1];
    float sum = x0 + x1, sq = x0 * x0 + x1 * x1;
#pragma unroll
    for (int m = 16; m > 0; m >>= 1) {
        sum += __shfl_xor(sum, m, 32);
        sq  += __shfl_xor(sq,  m, 32);
    }
    float mean = sum * (1.f / 64.f);
    float var  = sq * (1.f / 64.f) - mean * mean;
    float rs   = rsqrtf(var + EPSF);
    float y0 = fmaxf((x0 - mean) * rs * flnw[c0]     + flnb[c0],     0.f);
    float y1 = fmaxf((x1 - mean) * rs * flnw[c0 + 1] + flnb[c0 + 1], 0.f);
    v2h o; o[0] = (h16)y0; o[1] = (h16)y1;
    *(v2h*)(T + (size_t)e * 64 + c0) = o;
}

// agg[tgt[e]] += T[e] @ Wf + bf   -- wave per 16-edge tile, fused WMMA + atomic scatter.
// B fragments register-resident; wave grid-strides over edge tiles.
__global__ void __launch_bounds__(256, 1)
edge_gemm_scatter_kernel(const h16* __restrict__ T, const float* __restrict__ Wf,
                         const float* __restrict__ bf,
                         const long long* __restrict__ tgt,
                         float* __restrict__ agg, long long E) {
    __shared__ alignas(64) h16 wt[64 * 64];
    __shared__ float bl[64];
    int tid = threadIdx.x;
    for (int i = tid; i < 64 * 64; i += 256) {
        int k = i >> 6, n = i & 63;
        wt[n * 64 + k] = (h16)Wf[i];
    }
    if (tid < 64) bl[tid] = bf[tid];
    __syncthreads();

    int lane = tid & 31, wv = tid >> 5;

    v16h bfrag[2][4];
#pragma unroll
    for (int kc = 0; kc < 2; ++kc)
#pragma unroll
        for (int nt = 0; nt < 4; ++nt)
            bfrag[kc][nt] = load_b_frag(wt, 64, nt * 16, kc * 32, lane);

    int n_ = lane & 15, hi = lane >> 4;
    float bb[4];
#pragma unroll
    for (int nt = 0; nt < 4; ++nt) bb[nt] = bl[nt * 16 + n_];

    long long tiles = E >> 4;
    for (long long tile = (long long)blockIdx.x * 8 + wv; tile < tiles;
         tile += (long long)gridDim.x * 8) {
        long long row0 = tile * 16;
        v8f acc[4] = {};
#pragma unroll
        for (int kc = 0; kc < 2; ++kc) {
            v16h a = load_a_frag(T, 64, row0, kc * 32, lane);
#pragma unroll
            for (int nt = 0; nt < 4; ++nt) {
                acc[nt] = __builtin_amdgcn_wmma_f32_16x16x32_f16(
                    false, a, false, bfrag[kc][nt], (short)0, acc[nt], false, false);
            }
        }
        long long tg[8];
#pragma unroll
        for (int r = 0; r < 8; ++r) tg[r] = tgt[row0 + r + 8 * hi];
#pragma unroll
        for (int nt = 0; nt < 4; ++nt) {
            int n = nt * 16 + n_;
#pragma unroll
            for (int r = 0; r < 8; ++r) {
                atomicAdd(&agg[(size_t)tg[r] * 64 + n], acc[nt][r] + bb[nt]);
            }
        }
    }
}

// U[n] = [ layernorm(agg[n]; pln), right[n] ]  (f16, width 128). Wave per node.
__global__ void post_concat_kernel(const float* __restrict__ agg, const h16* __restrict__ right,
                                   const float* __restrict__ plnw, const float* __restrict__ plnb,
                                   h16* __restrict__ U, int N) {
    int tid = threadIdx.x, lane = tid & 31, wv = tid >> 5;
    int n = blockIdx.x * 8 + wv;
    if (n >= N) return;
    int c0 = lane * 2;
    float x0 = agg[(size_t)n * 64 + c0];
    float x1 = agg[(size_t)n * 64 + c0 + 1];
    float sum = x0 + x1, sq = x0 * x0 + x1 * x1;
#pragma unroll
    for (int m = 16; m > 0; m >>= 1) {
        sum += __shfl_xor(sum, m, 32);
        sq  += __shfl_xor(sq,  m, 32);
    }
    float mean = sum * (1.f / 64.f);
    float var  = sq * (1.f / 64.f) - mean * mean;
    float rs   = rsqrtf(var + EPSF);
    float y0 = (x0 - mean) * rs * plnw[c0]     + plnb[c0];
    float y1 = (x1 - mean) * rs * plnw[c0 + 1] + plnb[c0 + 1];
    v2h o; o[0] = (h16)y0; o[1] = (h16)y1;
    *(v2h*)(U + (size_t)n * 128 + c0) = o;
    *(v2h*)(U + (size_t)n * 128 + 64 + c0) = *(const v2h*)(right + (size_t)n * 64 + c0);
}

// out[n] = h[n] . Wh2 + bh2
__global__ void head_out_kernel(const h16* __restrict__ h, const float* __restrict__ W2,
                                const float* __restrict__ b2, float* __restrict__ out, int N) {
    int n = blockIdx.x * blockDim.x + threadIdx.x;
    if (n >= N) return;
    float acc = b2[0];
    const h16* hr = h + (size_t)n * 64;
#pragma unroll
    for (int k = 0; k < 64; ++k) acc += (float)hr[k] * W2[k];
    out[n] = acc;
}

// ---------------- host launch ----------------

struct ConvP {
    const float *Wl, *bl, *Wr, *flnw, *flnb, *Wf, *bf, *plnw, *plnb, *Wo1, *bo1, *Wo2, *bo2;
};

static inline int gemm_blocks(int rows) {
    int tiles = (rows + 15) / 16;
    int b = (tiles + 7) / 8;
    return b < 640 ? b : 640;      // cap so waves iterate several tiles (amortize B-frag setup)
}

extern "C" void kernel_launch(void* const* d_in, const int* in_sizes, int n_in,
                              void* d_out, int out_size, void* d_ws, size_t ws_size,
                              hipStream_t stream) {
    const float*     cons_x = (const float*)d_in[0];
    const long long* eidx   = (const long long*)d_in[1];   // int64 per reference
    const float*     var_x  = (const float*)d_in[3];

    const int Nc = in_sizes[0] / 5;
    const int Nv = in_sizes[3] / 19;
    const long long E = in_sizes[2];                        // edge_features count == E

    auto P = [&](int i) { return (const float*)d_in[i]; };
    // params flat order (insertion order of nested dicts):
    // cons_emb: 4 ln_w, 5 ln_b, 6 W1, 7 b1, 8 W2, 9 b2
    // var_emb : 10..15
    // v2c     : 16..28, c2v: 29..41, head: 42..45
    ConvP v2c = {P(16),P(17),P(18),P(19),P(20),P(21),P(22),P(23),P(24),P(25),P(26),P(27),P(28)};
    ConvP c2v = {P(29),P(30),P(31),P(32),P(33),P(34),P(35),P(36),P(37),P(38),P(39),P(40),P(41)};

    // ---- workspace bump allocator ----
    char* base = (char*)d_ws;
    size_t off = 0;
    auto alloc = [&](size_t bytes) {
        void* p = base + off;
        off = (off + bytes + 255) & ~(size_t)255;
        return p;
    };
    const int Nmax = Nc > Nv ? Nc : Nv;
    h16*   c_h    = (h16*)alloc((size_t)Nc   * 64 * 2);
    h16*   v_h    = (h16*)alloc((size_t)Nv   * 64 * 2);
    h16*   scr_h  = (h16*)alloc((size_t)Nmax * 64 * 2);
    h16*   hi_h   = (h16*)alloc((size_t)Nmax * 64 * 2);
    h16*   hj_h   = (h16*)alloc((size_t)Nmax * 64 * 2);
    h16*   cnew_h = (h16*)alloc((size_t)Nc   * 64 * 2);
    h16*   vnew_h = (h16*)alloc((size_t)Nv   * 64 * 2);
    float* agg    = (float*)alloc((size_t)Nmax * 64 * 4);
    h16*   U      = (h16*)alloc((size_t)Nmax * 128 * 2);
    h16*   T      = (h16*)alloc((size_t)E    * 64 * 2);

    const long long* e_cons = eidx;        // row 0: constraint index per edge
    const long long* e_var  = eidx + E;    // row 1: variable index per edge

    // ---- embeddings ----
    embed_front_kernel<<<(Nc * 64 + 255) / 256, 256, 0, stream>>>(
        cons_x, P(4), P(5), P(6), P(7), scr_h, Nc, 5);
    gemm_tile_kernel<64, 2><<<gemm_blocks(Nc), 256, 0, stream>>>(scr_h, P(8), P(9), c_h, Nc);

    embed_front_kernel<<<(Nv * 64 + 255) / 256, 256, 0, stream>>>(
        var_x, P(10), P(11), P(12), P(13), scr_h, Nv, 19);
    gemm_tile_kernel<64, 2><<<gemm_blocks(Nv), 256, 0, stream>>>(scr_h, P(14), P(15), v_h, Nv);

    // ---- one bipartite conv ----
    auto run_conv = [&](const h16* left_h, int Nl, const h16* right_h, int Nr,
                        const long long* tgtI, const long long* srcI,
                        const ConvP& cp, h16* out_h) {
        gemm_tile_kernel<64, 0><<<gemm_blocks(Nr), 256, 0, stream>>>(right_h, cp.Wl, cp.bl, hi_h, Nr);
        gemm_tile_kernel<64, 0><<<gemm_blocks(Nl), 256, 0, stream>>>(left_h,  cp.Wr, nullptr, hj_h, Nl);
        long long aggN = (long long)Nr * 64;
        zero_f32_kernel<<<(int)((aggN + 255) / 256), 256, 0, stream>>>(agg, aggN);
        int eb = (int)((E + 7) / 8);
        edge_prep_kernel<<<eb, 256, 0, stream>>>(hi_h, hj_h, tgtI, srcI, cp.flnw, cp.flnb, T, E);
        long long etiles = E / 16;
        int tb = (int)((etiles + 7) / 8);
        if (tb > 640) tb = 640;
        edge_gemm_scatter_kernel<<<tb, 256, 0, stream>>>(T, cp.Wf, cp.bf, tgtI, agg, E);
        post_concat_kernel<<<(Nr + 7) / 8, 256, 0, stream>>>(agg, right_h, cp.plnw, cp.plnb, U, Nr);
        gemm_tile_kernel<128, 1><<<gemm_blocks(Nr), 256, 0, stream>>>(U, cp.Wo1, cp.bo1, scr_h, Nr);
        gemm_tile_kernel<64, 0><<<gemm_blocks(Nr), 256, 0, stream>>>(scr_h, cp.Wo2, cp.bo2, out_h, Nr);
    };

    // v2c: left = vars, right = cons; tgt = cons index, src = var index
    run_conv(v_h, Nv, c_h, Nc, e_cons, e_var, v2c, cnew_h);
    // c2v: left = new cons, right = vars; tgt = var index, src = cons index
    run_conv(cnew_h, Nc, v_h, Nv, e_var, e_cons, c2v, vnew_h);

    // ---- head ----
    gemm_tile_kernel<64, 2><<<gemm_blocks(Nv), 256, 0, stream>>>(vnew_h, P(42), P(43), scr_h, Nv);
    head_out_kernel<<<(Nv + 255) / 256, 256, 0, stream>>>(scr_h, P(44), P(45), (float*)d_out, Nv);
}
